// LinearAngularAttention_37984690766448
// MI455X (gfx1250) — matrix operand order repaired and checked by reading, not compiled, not run on hardware
//
#include <hip/hip_runtime.h>

// Problem constants (match reference)
#define BB 16
#define SS 4096
#define DD 768
#define HH 12
#define HD 64
#define KQ 2304   // 3*DD
#define KSZ 9

typedef __attribute__((ext_vector_type(16))) __bf16 v16bf;
typedef __attribute__((ext_vector_type(8)))  float  v8f;
typedef __attribute__((ext_vector_type(4)))  int    v4i;

#define AS1 __attribute__((address_space(1)))
#define AS3 __attribute__((address_space(3)))

#if __has_builtin(__builtin_amdgcn_global_load_async_to_lds_b128) && \
    __has_builtin(__builtin_amdgcn_s_wait_asynccnt)
#define HAVE_ASYNC_LDS 1
#endif

static __device__ __forceinline__ v8f wmma_bf16(v16bf a, v16bf b, v8f c) {
  return __builtin_amdgcn_wmma_f32_16x16x32_bf16(false, a, false, b, (short)0, c, false, false);
}

// ---------- fragment loaders (CDNA5 wave32 WMMA VGPR layouts, ISA 7.12.2) ----------

// A (16x32 bf16) from row-major bf16: lane m = lane&15,
// VGPR i holds K pair k = (i>>2)*16 + ((lane<16)?0:8) + (i&3)*2
static __device__ __forceinline__ v16bf frag_A_bf16(const __bf16* A, long lda,
                                                    long m0, int k0, int lane) {
  long m = m0 + (lane & 15);
  int kb = (lane < 16) ? 0 : 8;
  const __bf16* p = A + m * lda + k0 + kb;
  union { uint4 u[2]; v16bf v; } u;
  u.u[0] = *(const uint4*)(p);
  u.u[1] = *(const uint4*)(p + 16);
  return u.v;
}

// A (16x32) from row-major f32, converted to bf16 in-register
static __device__ __forceinline__ v16bf frag_A_f32(const float* A, long lda,
                                                   long m0, int k0, int lane) {
  long m = m0 + (lane & 15);
  int kb = (lane < 16) ? 0 : 8;
  const float* p0 = A + m * lda + k0 + kb;
  const float* p1 = p0 + 16;
  float4 f0 = *(const float4*)(p0);
  float4 f1 = *(const float4*)(p0 + 4);
  float4 f2 = *(const float4*)(p1);
  float4 f3 = *(const float4*)(p1 + 4);
  v16bf a;
  a[0]=(__bf16)f0.x; a[1]=(__bf16)f0.y; a[2]=(__bf16)f0.z; a[3]=(__bf16)f0.w;
  a[4]=(__bf16)f1.x; a[5]=(__bf16)f1.y; a[6]=(__bf16)f1.z; a[7]=(__bf16)f1.w;
  a[8]=(__bf16)f2.x; a[9]=(__bf16)f2.y; a[10]=(__bf16)f2.z; a[11]=(__bf16)f2.w;
  a[12]=(__bf16)f3.x; a[13]=(__bf16)f3.y; a[14]=(__bf16)f3.z; a[15]=(__bf16)f3.w;
  return a;
}

// B (32x16) where B[k][n] = W[n][k], W row-major [N, ldw] bf16 (weight matrices)
static __device__ __forceinline__ v16bf frag_B_WT(const __bf16* W, long ldw,
                                                  int n0, int k0, int lane) {
  int n = n0 + (lane & 15);
  int kk = k0 + ((lane < 16) ? 0 : 8);
  const __bf16* p = W + (long)n * ldw + kk;
  union { uint4 u[2]; v16bf v; } u;
  u.u[0] = *(const uint4*)(p);
  u.u[1] = *(const uint4*)(p + 16);
  return u.v;
}

// B (32x16) from a row-major bf16 tile in LDS: B[k][n] = T[k*ld + n], element j at k = k0+kb+j+(j>=8?8:0)
static __device__ __forceinline__ v16bf frag_B_lds(const __bf16* T, int ld,
                                                   int n0, int k0, int lane) {
  int n = n0 + (lane & 15);
  int kk = k0 + ((lane < 16) ? 0 : 8);
  v16bf b;
#pragma unroll
  for (int j = 0; j < 16; ++j) b[j] = T[(kk + j + ((j >= 8) ? 8 : 0)) * ld + n];
  return b;
}

// ---------------- K0a: weight fp32 -> bf16 ; K0b: zero f32 ----------------
__global__ void cvt_f32_bf16_kernel(const float* __restrict__ src,
                                    __bf16* __restrict__ dst, long n) {
  long i = (long)blockIdx.x * blockDim.x + threadIdx.x;
  long stride = (long)gridDim.x * blockDim.x;
  for (; i < n; i += stride) dst[i] = (__bf16)src[i];
}

__global__ void zero_f32_kernel(float* __restrict__ dst, long n) {
  long i = (long)blockIdx.x * blockDim.x + threadIdx.x;
  if (i < n) dst[i] = 0.f;
}

// ---------------- K1: QKV GEMM + bias + q/k row L2-norm + layout shuffles ----------
// One wave computes a 64(M) x 64(N) tile: B fragments loaded once per k-step and
// reused across 4 row sub-tiles (4x less L2 weight traffic, 16 indep wmma chains).
// 64 N-cols == one head's full d, so the row norm for q/k stays wave-local.
__global__ __launch_bounds__(256) void qkv_gemm_kernel(
    const float* __restrict__ x,      // [B*S, 768]
    const __bf16* __restrict__ wq,    // [2304, 768] bf16
    const float* __restrict__ bq,     // [2304]
    __bf16* __restrict__ q_bf,        // [B,H,S,64]
    __bf16* __restrict__ kT_bf,       // [B,H,64,S]
    __bf16* __restrict__ v_bf,        // [B,H,S,64]
    float*  __restrict__ v_f32) {     // [B,H,S,64]  (lives in d_out scratch)
  __shared__ float red[8][16][64];
  const int lane  = threadIdx.x & 31;
  const int wslot = threadIdx.x >> 5;
  const int wid   = blockIdx.x * 8 + wslot;
  const int mt = wid / 36;                 // 1024 row-tiles of 64
  const int ch = wid % 36;                 // 64-col chunk
  const long m0 = (long)mt * 64;
  const int  n0 = ch * 64;

  v8f c[4][4] = {};
#pragma unroll 1
  for (int k0 = 0; k0 < DD; k0 += 32) {
    if (k0 + 64 < DD)
      __builtin_prefetch(x + (m0 + (lane & 15)) * DD + k0 + 64, 0, 1);
    v16bf bfr[4];
#pragma unroll
    for (int t = 0; t < 4; ++t)
      bfr[t] = frag_B_WT(wq, DD, n0 + t * 16, k0, lane);
#pragma unroll
    for (int ms = 0; ms < 4; ++ms) {
      v16bf a = frag_A_f32(x, DD, m0 + ms * 16, k0, lane);
#pragma unroll
      for (int t = 0; t < 4; ++t) c[ms][t] = wmma_bf16(a, bfr[t], c[ms][t]);
    }
  }

  const int nlo   = lane & 15;
  const int mbase = (lane < 16) ? 0 : 8;
  const int which = ch / 12;               // 0=q 1=k 2=v
  const int h     = ch % 12;
  const long b    = m0 / SS;
  const long bh   = b * HH + h;
  float (*tile)[64] = red[wslot];

#pragma unroll 1
  for (int ms = 0; ms < 4; ++ms) {
    const long sb = (m0 % SS) + ms * 16;
#pragma unroll
    for (int t = 0; t < 4; ++t) {
      float bias = bq[n0 + t * 16 + nlo];
#pragma unroll
      for (int r = 0; r < 8; ++r)
        tile[mbase + r][t * 16 + nlo] = c[ms][t][r] + bias;
    }
    // wave-local LDS; DS ops from one wave complete in order -> no barrier needed
    if (which == 2) {                      // v: fp32 + bf16 copies, no norm
      for (int i = lane; i < 1024; i += 32) {
        int r = i >> 6, e = i & 63;
        float val = tile[r][e];
        long idx = (bh * SS + sb + r) * HD + e;
        v_f32[idx] = val;
        v_bf[idx]  = (__bf16)val;
      }
    } else if (lane < 16) {                // q / k: row L2 normalize
      int r = lane;
      float ss = 0.f;
#pragma unroll
      for (int e = 0; e < 64; ++e) { float v = tile[r][e]; ss += v * v; }
      float inv = rsqrtf(ss);
      long s = sb + r;
      if (which == 0) {
        __bf16* dst = q_bf + (bh * SS + s) * HD;
#pragma unroll
        for (int e = 0; e < 64; ++e) dst[e] = (__bf16)(tile[r][e] * inv);
      } else {
#pragma unroll
        for (int e = 0; e < 64; ++e)
          kT_bf[(bh * HD + e) * SS + s] = (__bf16)(tile[r][e] * inv);
      }
    }
  }
}

// -------- K2: kv[b,h] += k^T v over a K(seq) segment; split-K with f32 atomics ----
// v tiles staged to LDS; CDNA5 async global->LDS DMA with double buffering when the
// toolchain exposes the builtins, synchronous LDS copy otherwise.
#define KVSPLIT 8
#define SEGLEN  (SS / KVSPLIT)   // 512
__global__ __launch_bounds__(128) void kv_gemm_kernel(
    const __bf16* __restrict__ kT,    // [B*H, 64, S]
    const __bf16* __restrict__ vbf,   // [B*H, S, 64]
    float* __restrict__ kv_acc) {     // [B*H, 64, 64] f32 (pre-zeroed)
  __shared__ __bf16 vt[2][32 * 64];
  const int bh   = blockIdx.x / KVSPLIT;
  const int seg  = blockIdx.x % KVSPLIT;
  const int lane = threadIdx.x & 31;
  const int wv   = threadIdx.x >> 5;   // wave -> 16-row strip of D
  const int d0   = wv * 16;
  const int sbeg = seg * SEGLEN;
  const __bf16* kTb = kT + (long)bh * HD * SS;
  const __bf16* vb  = vbf + (long)bh * SS * HD;

  v8f c[4] = {};
#ifdef HAVE_ASYNC_LDS
  for (int i = threadIdx.x; i < 256; i += 128)
    __builtin_amdgcn_global_load_async_to_lds_b128(
        (AS1 v4i*)((v4i*)(vb + (long)sbeg * HD) + i),
        (AS3 v4i*)((v4i*)vt[0] + i), 0, 0);
#endif
#pragma unroll 1
  for (int s0 = sbeg; s0 < sbeg + SEGLEN; s0 += 32) {
    const int cur = ((s0 - sbeg) >> 5) & 1;
#ifdef HAVE_ASYNC_LDS
    __builtin_amdgcn_s_wait_asynccnt(0);   // my portion of tile `cur` landed
    __syncthreads();                       // everyone's portion landed, prev reads done
    if (s0 + 32 < sbeg + SEGLEN)
      for (int i = threadIdx.x; i < 256; i += 128)
        __builtin_amdgcn_global_load_async_to_lds_b128(
            (AS1 v4i*)((v4i*)(vb + (long)(s0 + 32) * HD) + i),
            (AS3 v4i*)((v4i*)vt[cur ^ 1] + i), 0, 0);
#else
    __syncthreads();
    {
      uint4* dst = (uint4*)vt[cur];
      const uint4* src = (const uint4*)(vb + (long)s0 * HD);
      for (int i = threadIdx.x; i < 256; i += 128) dst[i] = src[i];
    }
    __syncthreads();
#endif
    v16bf a = frag_A_bf16(kTb, SS, d0, s0, lane);
#pragma unroll
    for (int t = 0; t < 4; ++t) {
      v16bf b = frag_B_lds(vt[cur], HD, t * 16, 0, lane);
      c[t] = wmma_bf16(a, b, c[t]);
    }
  }
  const int nlo = lane & 15;
  const int mbase = (lane < 16) ? 0 : 8;
#pragma unroll
  for (int t = 0; t < 4; ++t)
#pragma unroll
    for (int r = 0; r < 8; ++r)
      atomicAdd(&kv_acc[(long)bh * 4096 + (long)(d0 + mbase + r) * HD + t * 16 + nlo],
                c[t][r]);
}

// ---------------- K2b: kv f32 -> bf16 with 1/pi fold ----------------
__global__ void kv_scale_kernel(const float* __restrict__ acc,
                                __bf16* __restrict__ kv_bf, long n) {
  const float invpi = 0.318309886183790672f;
  long i = (long)blockIdx.x * blockDim.x + threadIdx.x;
  if (i < n) kv_bf[i] = (__bf16)(acc[i] * invpi);
}

// --------- K3: out = normalize(0.5 v + q@(kv/pi)) + dwconv(v); write [B,S,H*d] ----
__global__ __launch_bounds__(256) void attn_out_kernel(
    const __bf16* __restrict__ qbf,   // [B*H, S, 64]
    const __bf16* __restrict__ kvbf,  // [B*H, 64, 64] (pre-scaled 1/pi)
    const float*  __restrict__ vf32,  // [B*H, S, 64]
    const float*  __restrict__ wdc,   // [H, 9]
    __bf16* __restrict__ outbf) {     // [B, S, 768] bf16 (head-major cols)
  __shared__ __bf16 kvs[64 * 64];
  __shared__ float  wconv[KSZ];
  __shared__ float  tile[8][16][64];
  const int bh = blockIdx.x >> 5;
  const int mg = blockIdx.x & 31;
  const int h  = bh % HH;
  const int lane = threadIdx.x & 31;
  const int wv   = threadIdx.x >> 5;

  {
    uint4* dst = (uint4*)kvs;
    const uint4* src = (const uint4*)(kvbf + (long)bh * 4096);
    for (int i = threadIdx.x; i < 512; i += 256) dst[i] = src[i];
    if (threadIdx.x < KSZ) wconv[threadIdx.x] = wdc[h * KSZ + threadIdx.x];
  }
  __syncthreads();

  const long s0 = (long)(mg * 8 + wv) * 16;
  const __bf16* qb = qbf + (long)bh * SS * HD;
  const float*  vb = vf32 + (long)bh * SS * HD;

  v8f c[4] = {};
#pragma unroll
  for (int k0 = 0; k0 < HD; k0 += 32) {
    v16bf a = frag_A_bf16(qb, HD, s0, k0, lane);
#pragma unroll
    for (int t = 0; t < 4; ++t) {
      v16bf b = frag_B_lds(kvs, HD, t * 16, k0, lane);
      c[t] = wmma_bf16(a, b, c[t]);
    }
  }

  const int nlo = lane & 15;
  const int mbase = (lane < 16) ? 0 : 8;
#pragma unroll
  for (int t = 0; t < 4; ++t)
#pragma unroll
    for (int r = 0; r < 8; ++r)
      tile[wv][mbase + r][t * 16 + nlo] =
          c[t][r] + 0.5f * vb[(s0 + mbase + r) * HD + t * 16 + nlo];

  if (lane < 16) {           // per-row L2 normalize (wave-local LDS)
    float ss = 0.f;
#pragma unroll
    for (int e = 0; e < 64; ++e) { float v = tile[wv][lane][e]; ss += v * v; }
    float inv = rsqrtf(ss);
#pragma unroll
    for (int e = 0; e < 64; ++e) tile[wv][lane][e] *= inv;
  }

  const long bq = bh / HH;
  for (int i = lane; i < 1024; i += 32) {   // + depthwise conv, write fused-transpose
    int r = i >> 6, e = i & 63;
    long s = s0 + r;
    float acc = 0.f;
#pragma unroll
    for (int t = 0; t < KSZ; ++t) {
      long sp = s + t - (KSZ / 2);
      if (sp >= 0 && sp < SS) acc += wconv[t] * vb[sp * HD + e];
    }
    float val = tile[wv][r][e] + acc;
    outbf[(bq * SS + s) * DD + h * HD + e] = (__bf16)val;
  }
}

// ---------------- K4: final projection y = out @ w_proj^T + b_proj ----------------
__global__ __launch_bounds__(256) void proj_gemm_kernel(
    const __bf16* __restrict__ A,     // [B*S, 768] bf16
    const __bf16* __restrict__ wp,    // [768, 768] bf16
    const float*  __restrict__ bp,    // [768]
    float* __restrict__ y) {          // [B*S, 768] f32
  const int lane  = threadIdx.x & 31;
  const int wid   = blockIdx.x * 8 + (threadIdx.x >> 5);
  const int mt = wid / 12;
  const int ch = wid % 12;
  const long m0 = (long)mt * 64;
  const int  n0 = ch * 64;

  v8f c[4][4] = {};
#pragma unroll 1
  for (int k0 = 0; k0 < DD; k0 += 32) {
    if (k0 + 64 < DD)
      __builtin_prefetch(A + (m0 + (lane & 15)) * DD + k0 + 64, 0, 1);
    v16bf bfr[4];
#pragma unroll
    for (int t = 0; t < 4; ++t)
      bfr[t] = frag_B_WT(wp, DD, n0 + t * 16, k0, lane);
#pragma unroll
    for (int ms = 0; ms < 4; ++ms) {
      v16bf a = frag_A_bf16(A, DD, m0 + ms * 16, k0, lane);
#pragma unroll
      for (int t = 0; t < 4; ++t) c[ms][t] = wmma_bf16(a, bfr[t], c[ms][t]);
    }
  }
  const int nlo = lane & 15;
  const int mbase = (lane < 16) ? 0 : 8;
#pragma unroll
  for (int ms = 0; ms < 4; ++ms)
#pragma unroll
    for (int t = 0; t < 4; ++t) {
      float bias = bp[n0 + t * 16 + nlo];
#pragma unroll
      for (int r = 0; r < 8; ++r)
        y[(m0 + ms * 16 + mbase + r) * DD + n0 + t * 16 + nlo] = c[ms][t][r] + bias;
    }
}

// ---------------------------------- launcher ----------------------------------
extern "C" void kernel_launch(void* const* d_in, const int* in_sizes, int n_in,
                              void* d_out, int out_size, void* d_ws, size_t ws_size,
                              hipStream_t stream) {
  const float* x   = (const float*)d_in[0];
  const float* wq  = (const float*)d_in[1];
  const float* bq  = (const float*)d_in[2];
  const float* wp  = (const float*)d_in[3];
  const float* bp  = (const float*)d_in[4];
  const float* wdc = (const float*)d_in[5];

  char* ws = (char*)d_ws;
  size_t off = 0;
  auto alloc = [&](size_t bytes) -> void* {
    void* p = ws + off;
    off = (off + bytes + 255) & ~(size_t)255;
    return p;
  };
  const long NTOK = (long)BB * SS;                 // 65536
  const long NKV  = (long)BB * HH * HD * HD;       // 786432
  __bf16* wq_bf  = (__bf16*)alloc((size_t)KQ * DD * 2);
  __bf16* wp_bf  = (__bf16*)alloc((size_t)DD * DD * 2);
  __bf16* q_bf   = (__bf16*)alloc((size_t)NTOK * DD * 2);
  __bf16* kT_bf  = (__bf16*)alloc((size_t)NTOK * DD * 2);
  __bf16* v_bf   = (__bf16*)alloc((size_t)NTOK * DD * 2);
  float*  kv_acc = (float*)alloc((size_t)NKV * 4);
  __bf16* kv_bf  = (__bf16*)alloc((size_t)NKV * 2);
  __bf16* out_bf = (__bf16*)alloc((size_t)NTOK * DD * 2);
  float*  v_f32  = (float*)d_out;   // scratch: dead before K4 overwrites d_out

  cvt_f32_bf16_kernel<<<512, 256, 0, stream>>>(wq, wq_bf, (long)KQ * DD);
  cvt_f32_bf16_kernel<<<256, 256, 0, stream>>>(wp, wp_bf, (long)DD * DD);
  zero_f32_kernel<<<(int)(NKV / 256), 256, 0, stream>>>(kv_acc, NKV);

  // K1: 1024 row-tiles(64) * 36 chunks = 36864 waves / 8 per block
  qkv_gemm_kernel<<<4608, 256, 0, stream>>>(x, wq_bf, bq, q_bf, kT_bf, v_bf, v_f32);

  // K2: (b,h) x 8 seq-segments, f32 atomic reduction
  kv_gemm_kernel<<<BB * HH * KVSPLIT, 128, 0, stream>>>(kT_bf, v_bf, kv_acc);
  kv_scale_kernel<<<(int)(NKV / 256), 256, 0, stream>>>(kv_acc, kv_bf, NKV);

  // K3: 192 (b,h) * 32 row-groups
  attn_out_kernel<<<BB * HH * 32, 256, 0, stream>>>(q_bf, kv_bf, v_f32, wdc, out_bf);

  // K4: 1024 row-tiles(64) * 12 chunks = 12288 waves / 8 per block
  proj_gemm_kernel<<<1536, 256, 0, stream>>>(out_bf, wp_bf, bp, (float*)d_out);

  (void)in_sizes; (void)n_in; (void)out_size; (void)ws_size;
}
